// RegressionHead_49830210568640
// MI455X (gfx1250) — matrix operand order repaired and checked by reading, not compiled, not run on hardware
//
#include <hip/hip_runtime.h>

typedef __attribute__((ext_vector_type(16))) _Float16 v16h;
typedef __attribute__((ext_vector_type(2)))  _Float16 v2h;
typedef __attribute__((ext_vector_type(8)))  float    v8f;
typedef __attribute__((ext_vector_type(4)))  float    v4f;

#define BB   16
#define NN   65536
#define DD   256
#define HH   64
#define KTOP 256
#define EPSF 1e-5f

__device__ __forceinline__ unsigned key_of(float f) {
  unsigned u = __float_as_uint(f);
  return (u & 0x80000000u) ? ~u : (u | 0x80000000u);
}
__device__ __forceinline__ float key_inv(unsigned k) {
  return __uint_as_float((k & 0x80000000u) ? (k ^ 0x80000000u) : ~k);
}

// ---------------------------------------------------------------------------
// Kernel 0: query projection (exact fp32), fold r_g1/r_b1 into W and bias,
// build WMMA B-fragment-ordered f16 copy of r_W, per-batch g2*q/sqrt(H) etc.
// ---------------------------------------------------------------------------
__global__ void prep_kernel(
    const float* __restrict__ query,
    const float* __restrict__ qg1, const float* __restrict__ qb1,
    const float* __restrict__ qW,  const float* __restrict__ qbv,
    const float* __restrict__ qg2, const float* __restrict__ qb2,
    const float* __restrict__ rg1, const float* __restrict__ rb1,
    const float* __restrict__ rW,  const float* __restrict__ rbv,
    const float* __restrict__ rg2, const float* __restrict__ rb2,
    float* __restrict__ qbuf, float* __restrict__ g2q,
    float* __restrict__ b2qs, float* __restrict__ bias2,
    _Float16* __restrict__ wfrag)
{
  const int t = threadIdx.x;          // 256 threads
  const int lane = t & 31, wave = t >> 5;
  __shared__ float s_xn[DD];
  __shared__ float s_ps[8], s_pq[8];

  for (int b = 0; b < BB; ++b) {
    // LayerNorm over D=256 (one element per thread)
    float x = query[b * DD + t];
    float s = x, sq = x * x;
    #pragma unroll
    for (int m = 1; m < 32; m <<= 1) { s += __shfl_xor(s, m, 32); sq += __shfl_xor(sq, m, 32); }
    if (lane == 0) { s_ps[wave] = s; s_pq[wave] = sq; }
    __syncthreads();
    float ts = 0.f, tq = 0.f;
    #pragma unroll
    for (int w = 0; w < 8; ++w) { ts += s_ps[w]; tq += s_pq[w]; }
    const float mu = ts * (1.f / DD);
    const float rstd = rsqrtf(tq * (1.f / DD) - mu * mu + EPSF);
    s_xn[t] = (x - mu) * rstd * qg1[t] + qb1[t];
    __syncthreads();
    // Linear D->H (threads 0..63)
    float hval = 0.f;
    if (t < HH) {
      hval = qbv[t];
      for (int d = 0; d < DD; ++d) hval += s_xn[d] * qW[t * DD + d];
    }
    __syncthreads();
    // LayerNorm over H=64 (waves 0,1 only)
    float s2 = (t < HH) ? hval : 0.f;
    float q2 = s2 * s2;
    #pragma unroll
    for (int m = 1; m < 32; m <<= 1) { s2 += __shfl_xor(s2, m, 32); q2 += __shfl_xor(q2, m, 32); }
    if (lane == 0) { s_ps[wave] = s2; s_pq[wave] = q2; }
    __syncthreads();
    if (t < HH) {
      const float hs = s_ps[0] + s_ps[1];
      const float hq = s_pq[0] + s_pq[1];
      const float mu2 = hs * (1.f / HH);
      const float rstd2 = rsqrtf(hq * (1.f / HH) - mu2 * mu2 + EPSF);
      qbuf[b * HH + t] = (hval - mu2) * rstd2 * qg2[t] + qb2[t];
    }
    __syncthreads();
  }

  const float ish = 0.125f;   // 1/sqrt(64)
  for (int i = t; i < BB * HH; i += 256) {
    int j = i & (HH - 1);
    g2q[i] = rg2[j] * qbuf[i] * ish;
  }
  if (t < BB) {
    float acc = 0.f;
    for (int j = 0; j < HH; ++j) acc += rb2[j] * qbuf[t * HH + j];
    b2qs[t] = acc * ish;
  }
  if (t < HH) {
    float acc = rbv[t];
    for (int d = 0; d < DD; ++d) acc += rb1[d] * rW[t * DD + d];
    bias2[t] = acc;
  }
  // B-fragment swizzle of W' = r_W * r_g1 (f16):
// wfrag[kt][nt][lane][e], K=kt*32+8*(lane/16)+e for e<8, K=kt*32+16+8*(lane/16)+(e-8) for e>=8,
  // column N = nt*16 + lane%16.
  for (int i = t; i < 8 * 4 * 32 * 16; i += 256) {
    const int e  = i & 15;
    const int ln = (i >> 4) & 31;
    const int nt = (i >> 9) & 3;
    const int kt = i >> 11;
    const int j  = nt * 16 + (ln & 15);
    const int d  = kt * 32 + ((ln >> 4) << 3) + (e & 7) + ((e >> 3) << 4);
    wfrag[i] = (_Float16)(rW[j * DD + d] * rg1[d]);
  }
}

// ---------------------------------------------------------------------------
// Kernel 1: stream ref_repr once (non-temporal); fused LN(D) -> WMMA f16 GEMM
// (D->H) -> LN(H) -> score dot. 8 waves/block, 16 rows/wave/tile, 4 tiles.
// Row cached as packed f16 (64 VGPRs); B-fragments re-read from LDS per tile
// (opaque base offset defeats register hoisting) to keep VGPRs < 256.
// ---------------------------------------------------------------------------
__global__ __launch_bounds__(256) void proj_kernel(
    const float* __restrict__ ref,        // [B,N,D]
    const _Float16* __restrict__ wfrag,   // [8][4][32][16] f16
    const float* __restrict__ g2q,        // [B,H]
    const float* __restrict__ b2qs,       // [B]
    const float* __restrict__ bias2,      // [H]
    float* __restrict__ scores)           // [B,N]
{
  __shared__ __align__(32) _Float16 sW[8 * 4 * 32 * 16];
  __shared__ float s_g2q[HH];
  __shared__ float s_b2[HH];
  const int t = threadIdx.x;
  const int b = blockIdx.y;

  for (int i = t; i < 2048; i += 256)
    *(v4f*)&sW[i * 8] = *(const v4f*)&wfrag[i * 8];
  if (t < HH) { s_g2q[t] = g2q[b * HH + t]; s_b2[t] = bias2[t]; }
  __syncthreads();

  const float badd = b2qs[b];
  const int lane = t & 31;
  const int wave = t >> 5;
  const int hi = lane >> 4;   // half-wave id
  const int lr = lane & 15;   // row within tile for A / column for D
  const int c0 = hi * 8;      // A-fragment K offset for this half-wave

  float bias_l[4], g2q_l[4];
  #pragma unroll
  for (int nt = 0; nt < 4; ++nt) { bias_l[nt] = s_b2[nt * 16 + lr]; g2q_l[nt] = s_g2q[nt * 16 + lr]; }

  #pragma unroll 1
  for (int it = 0; it < 4; ++it) {
    const int rowbase = blockIdx.x * 512 + it * 128 + wave * 16;
    const float* rowp = ref + ((size_t)b * NN + rowbase + lr) * DD;

    // Opaque zero: compiler cannot prove the LDS B-fragment reads are
    // loop-invariant, so they are not hoisted into 256 VGPRs.
    int zro;
    asm volatile("s_mov_b32 %0, 0" : "=s"(zro));
    const _Float16* sWit = sW + zro;

    // Load this lane's half of the row (A-fragment order), f32 stats, pack f16
    v2h xh[64];
    float s = 0.f, sq = 0.f;
    #pragma unroll
    for (int kt = 0; kt < 8; ++kt) {
      const float* p = rowp + kt * 32 + c0;
      const v4f u0 = __builtin_nontemporal_load((const v4f*)(p));
      const v4f u1 = __builtin_nontemporal_load((const v4f*)(p + 4));
      const v4f u2 = __builtin_nontemporal_load((const v4f*)(p + 16));
      const v4f u3 = __builtin_nontemporal_load((const v4f*)(p + 20));
      #pragma unroll
      for (int j = 0; j < 4; ++j) { s += u0[j]; sq += u0[j] * u0[j]; }
      #pragma unroll
      for (int j = 0; j < 4; ++j) { s += u1[j]; sq += u1[j] * u1[j]; }
      #pragma unroll
      for (int j = 0; j < 4; ++j) { s += u2[j]; sq += u2[j] * u2[j]; }
      #pragma unroll
      for (int j = 0; j < 4; ++j) { s += u3[j]; sq += u3[j] * u3[j]; }
      xh[kt * 8 + 0] = (v2h){(_Float16)u0[0], (_Float16)u0[1]};
      xh[kt * 8 + 1] = (v2h){(_Float16)u0[2], (_Float16)u0[3]};
      xh[kt * 8 + 2] = (v2h){(_Float16)u1[0], (_Float16)u1[1]};
      xh[kt * 8 + 3] = (v2h){(_Float16)u1[2], (_Float16)u1[3]};
      xh[kt * 8 + 4] = (v2h){(_Float16)u2[0], (_Float16)u2[1]};
      xh[kt * 8 + 5] = (v2h){(_Float16)u2[2], (_Float16)u2[3]};
      xh[kt * 8 + 6] = (v2h){(_Float16)u3[0], (_Float16)u3[1]};
      xh[kt * 8 + 7] = (v2h){(_Float16)u3[2], (_Float16)u3[3]};
    }
    // LayerNorm stats over D=256 (pair lanes l and l+16 share a row)
    s  += __shfl_xor(s, 16, 32);
    sq += __shfl_xor(sq, 16, 32);
    const float mu = s * (1.f / 256.f);
    const float rstd = rsqrtf(sq * (1.f / 256.f) - mu * mu + EPSF);
    const _Float16 rsh = (_Float16)rstd;
    const _Float16 nbh = (_Float16)(-mu * rstd);
    const v2h rs2 = (v2h){rsh, rsh};
    const v2h nb2 = (v2h){nbh, nbh};

    // h[16x64] = LN(x) @ W'^T via 8 K-steps x 4 N-tiles of WMMA f16
    v8f acc[4] = {};
    #pragma unroll
    for (int kt = 0; kt < 8; ++kt) {
      v16h a;
      #pragma unroll
      for (int j = 0; j < 8; ++j) {
        const v2h t2 = xh[kt * 8 + j] * rs2 + nb2;   // v_pk_fma_f16
        a[2 * j]     = t2[0];
        a[2 * j + 1] = t2[1];
      }
      #pragma unroll
      for (int nt = 0; nt < 4; ++nt) {
        const v16h bf = *(const v16h*)&sWit[((kt * 4 + nt) * 32 + lane) * 16];
        acc[nt] = __builtin_amdgcn_wmma_f32_16x16x32_f16(
            false, a, false, bf, (short)0, acc[nt], false, false);
      }
    }
    // LayerNorm over H=64 per row (row M = r + 8*hi lives in one half-wave)
    float hs[8], hq[8];
    #pragma unroll
    for (int r = 0; r < 8; ++r) {
      const float a0 = acc[0][r] + bias_l[0];
      const float a1 = acc[1][r] + bias_l[1];
      const float a2 = acc[2][r] + bias_l[2];
      const float a3 = acc[3][r] + bias_l[3];
      hs[r] = a0 + a1 + a2 + a3;
      hq[r] = a0 * a0 + a1 * a1 + a2 * a2 + a3 * a3;
    }
    #pragma unroll
    for (int m = 1; m < 16; m <<= 1) {
      #pragma unroll
      for (int r = 0; r < 8; ++r) { hs[r] += __shfl_xor(hs[r], m, 32); hq[r] += __shfl_xor(hq[r], m, 32); }
    }
    float pr[8], rstd2[8];
    #pragma unroll
    for (int r = 0; r < 8; ++r) {
      const float mu2 = hs[r] * (1.f / 64.f);
      rstd2[r] = rsqrtf(hq[r] * (1.f / 64.f) - mu2 * mu2 + EPSF);
      float p = 0.f;
      #pragma unroll
      for (int nt = 0; nt < 4; ++nt) p += (acc[nt][r] + bias_l[nt] - mu2) * g2q_l[nt];
      pr[r] = p;
    }
    #pragma unroll
    for (int m = 1; m < 16; m <<= 1) {
      #pragma unroll
      for (int r = 0; r < 8; ++r) pr[r] += __shfl_xor(pr[r], m, 32);
    }
    if (lr == 0) {
      v4f o0, o1;
      #pragma unroll
      for (int r = 0; r < 4; ++r) o0[r] = pr[r] * rstd2[r] + badd;
      #pragma unroll
      for (int r = 0; r < 4; ++r) o1[r] = pr[r + 4] * rstd2[r + 4] + badd;
      float* op = scores + (size_t)b * NN + rowbase + hi * 8;
      *(v4f*)op = o0;
      *(v4f*)(op + 4) = o1;
    }
  }
}

// ---------------------------------------------------------------------------
// Kernel 2: per batch, exact radix-select of K-th largest score, then
// masked max-subtract softmax + weighted sum of ref_values. Deterministic.
// ---------------------------------------------------------------------------
__global__ __launch_bounds__(1024) void topk_kernel(
    const float* __restrict__ scores,
    const float* __restrict__ refv,
    const float* __restrict__ tau,
    float* __restrict__ out)
{
  const int b = blockIdx.x;
  const int t = threadIdx.x;          // 1024 threads, 64 scores each
  const int lane = t & 31, wave = t >> 5;
  const float* sc = scores + (size_t)b * NN;

  __shared__ int hist[256];
  __shared__ unsigned s_maxkey;
  __shared__ float s_wa[32], s_wb[32];

  unsigned prefix = 0;
  int kth = KTOP;
  for (int pass = 0; pass < 4; ++pass) {
    const int shift = 24 - pass * 8;
    if (t < 256) hist[t] = 0;
    if (t == 0 && pass == 0) s_maxkey = 0u;
    __syncthreads();
    unsigned lmax = 0u;
    for (int i = 0; i < 64; ++i) {
      const unsigned key = key_of(sc[t + i * 1024]);
      if (pass == 0) {
        lmax = lmax > key ? lmax : key;
        atomicAdd(&hist[key >> 24], 1);
      } else {
        if ((key >> (shift + 8)) == (prefix >> (shift + 8)))
          atomicAdd(&hist[(key >> shift) & 255], 1);
      }
    }
    if (pass == 0) {
      #pragma unroll
      for (int m = 1; m < 32; m <<= 1) {
        const unsigned o = (unsigned)__shfl_xor((int)lmax, m, 32);
        lmax = lmax > o ? lmax : o;
      }
      if (lane == 0) atomicMax(&s_maxkey, lmax);
    }
    __syncthreads();
    // All threads scan the (identical) histogram: deterministic & uniform
    int cum = 0, chosen = 0, newk = kth;
    for (int bin = 255; bin >= 0; --bin) {
      const int c = hist[bin];
      if (cum + c >= kth) { chosen = bin; newk = kth - cum; break; }
      cum += c;
    }
    kth = newk;
    prefix |= ((unsigned)chosen) << shift;
    __syncthreads();
  }
  const unsigned thresh = prefix;      // K-th largest key
  const float smax = key_inv(s_maxkey);
  const float itau = 1.f / tau[0];

  float se = 0.f, sv = 0.f;
  const float* rv = refv + (size_t)b * NN;
  for (int i = 0; i < 64; ++i) {
    const int n = t + i * 1024;
    const float f = sc[n];
    if (key_of(f) >= thresh) {
      const float e = __expf((f - smax) * itau);
      se += e;
      sv += e * rv[n];
    }
  }
  #pragma unroll
  for (int m = 1; m < 32; m <<= 1) { se += __shfl_xor(se, m, 32); sv += __shfl_xor(sv, m, 32); }
  if (lane == 0) { s_wa[wave] = se; s_wb[wave] = sv; }
  __syncthreads();
  if (t == 0) {
    float tse = 0.f, tsv = 0.f;
    for (int w = 0; w < 32; ++w) { tse += s_wa[w]; tsv += s_wb[w]; }
    out[b] = tsv / tse;
  }
}

// ---------------------------------------------------------------------------
extern "C" void kernel_launch(void* const* d_in, const int* in_sizes, int n_in,
                              void* d_out, int out_size, void* d_ws, size_t ws_size,
                              hipStream_t stream)
{
  (void)in_sizes; (void)n_in; (void)out_size; (void)ws_size;
  const float* query = (const float*)d_in[0];
  const float* ref   = (const float*)d_in[1];
  const float* refv  = (const float*)d_in[2];
  const float* tau   = (const float*)d_in[3];
  const float* qg1   = (const float*)d_in[4];
  const float* qb1   = (const float*)d_in[5];
  const float* qW    = (const float*)d_in[6];
  const float* qbv   = (const float*)d_in[7];
  const float* qg2   = (const float*)d_in[8];
  const float* qb2   = (const float*)d_in[9];
  const float* rg1   = (const float*)d_in[10];
  const float* rb1   = (const float*)d_in[11];
  const float* rW    = (const float*)d_in[12];
  const float* rbv   = (const float*)d_in[13];
  const float* rg2   = (const float*)d_in[14];
  const float* rb2   = (const float*)d_in[15];

  char* ws = (char*)d_ws;
  float*    qbuf   = (float*)(ws + 0);        // 4KB
  float*    g2q    = (float*)(ws + 4096);     // 4KB
  float*    b2qs   = (float*)(ws + 8192);     // 64B
  float*    bias2  = (float*)(ws + 8448);     // 256B
  _Float16* wfrag  = (_Float16*)(ws + 16384); // 32KB
  float*    scores = (float*)(ws + 65536);    // 4MB

  prep_kernel<<<1, 256, 0, stream>>>(query, qg1, qb1, qW, qbv, qg2, qb2,
                                     rg1, rb1, rW, rbv, rg2, rb2,
                                     qbuf, g2q, b2qs, bias2, wfrag);
  proj_kernel<<<dim3(128, 16), 256, 0, stream>>>(ref, wfrag, g2q, b2qs, bias2, scores);
  topk_kernel<<<16, 1024, 0, stream>>>(scores, refv, tau, (float*)d_out);
}
